// DGLJTNNEncoder_58274116272735
// MI455X (gfx1250) — compile-verified, matrix operand order checked by reading, not run on hardware
//
#include <hip/hip_runtime.h>
#include <hip/hip_bf16.h>

// ---------------------------------------------------------------------------
// DGL-JTNN tree-GRU encoder for gfx1250 (MI455X), wave32 + WMMA f16->f32.
// ---------------------------------------------------------------------------

#define HDIM 256              // hidden size H
#define LDSH 264              // padded LDS row stride (halves), 16B-aligned rows
#define NTREES 256
#define NPT 255
#define EPT 508
#define NNODES (NTREES * NPT) // 65280
#define NEDGES (NTREES * EPT) // 130048
#define NLEVELS 14

typedef __attribute__((ext_vector_type(16))) _Float16 v16h;
typedef __attribute__((ext_vector_type(8)))  _Float16 v8h;
typedef __attribute__((ext_vector_type(8)))  float    v8f;

__device__ __forceinline__ float sigmoidf_(float x) {
    return 1.0f / (1.0f + __expf(-x));
}

__device__ __forceinline__ v8f wmma_f16(v16h a, v16h b, v8f c) {
    // D(f32, 16x16) = A(16x32 f16) * B(32x16 f16) + C
    return __builtin_amdgcn_wmma_f32_16x16x32_f16(
        /*neg_a=*/false, a, /*neg_b=*/false, b,
        /*c_mod=*/(short)0, c, /*reuse_a=*/false, /*reuse_b=*/false);
}

// ISA 16-bit A/B fragment K-mapping: lane<16 holds K {0..7,16..23},
// lane>=16 holds K {8..15,24..31}; consecutive half pairs per VGPR.
__device__ __forceinline__ int kmap(int i, int l) {
    return ((i < 8) ? i : i + 8) + (((l & 31) < 16) ? 0 : 8);
}

// Load a 16x32 A-fragment row from LDS (this lane's row = l&15), ktile base in halves.
__device__ __forceinline__ v16h loadA(const _Float16* row, int kbase, int l) {
    const int sel = ((l & 31) < 16) ? 0 : 8;
    v8h lo = *(const v8h*)(row + kbase + sel);        // 16B LDS load
    v8h hi = *(const v8h*)(row + kbase + 16 + sel);   // 16B LDS load
    v16h a;
#pragma unroll
    for (int i = 0; i < 8; ++i) { a[i] = lo[i]; a[i + 8] = hi[i]; }
    return a;
}

// Pre-swizzled B fragment: one 32B contiguous load per lane.
__device__ __forceinline__ v16h loadB(const _Float16* wfrag, int kt, int nt, int l) {
    return *(const v16h*)(wfrag + ((((kt * 16) + nt) * 32 + (l & 31)) << 4));
}

// ---------------------------------------------------------------------------
// Weight conversion: f32 row-major [K,256] -> f16 fragment-major image so that
// loadB() is a single contiguous 32-byte load per lane.
// ---------------------------------------------------------------------------
__global__ void cvt_frag_kernel(const float* __restrict__ W, _Float16* __restrict__ out, int K) {
    long t = (long)blockIdx.x * blockDim.x + threadIdx.x;
    if (t >= (long)K * HDIM) return;
    int i  = (int)(t & 15);
    int l  = (int)((t >> 4) & 31);
    int nt = (int)((t >> 9) & 15);
    int kt = (int)(t >> 13);
    int k  = kt * 32 + kmap(i, l);
    int n  = nt * 16 + (l & 15);
    out[t] = (_Float16)W[(long)k * HDIM + n];
}

// x16[node][h] = (f16) emb[wid[node]][h]
__global__ void gather_x_kernel(const float* __restrict__ emb, const int* __restrict__ wid,
                                _Float16* __restrict__ x16) {
    long t = (long)blockIdx.x * 256 + threadIdx.x;
    int node = (int)(t >> 8);
    int h    = (int)(t & 255);
    x16[t] = (_Float16)emb[(long)wid[node] * HDIM + h];
}

__global__ void zerof_kernel(float* __restrict__ p, long n) {
    long t = (long)blockIdx.x * 256 + threadIdx.x;
    if (t < n) p[t] = 0.0f;
}

// Per-level predecessor reduction: s[j] += m[e]; rm[j] += r[e]*m[e]
// One block per list entry, 256 threads = H lanes.
__global__ void scatter_sm_kernel(const int* __restrict__ lgs, const int* __restrict__ lgd,
                                  const float* __restrict__ m_out, const _Float16* __restrict__ r16,
                                  float* __restrict__ s, float* __restrict__ rm) {
    int entry = blockIdx.x;
    int h = threadIdx.x;
    int e = lgs[entry];
    int j = lgd[entry];
    float mv = m_out[(long)e * HDIM + h];
    float rv = (float)r16[(long)e * HDIM + h];
    atomicAdd(&s[(long)j * HDIM + h], mv);
    atomicAdd(&rm[(long)j * HDIM + h], rv * mv);
}

// accum[dst[e]] += m[e]
__global__ void scatter_accum_kernel(const int* __restrict__ dstI, const float* __restrict__ m_out,
                                     float* __restrict__ accum) {
    long e = blockIdx.x;
    int h = threadIdx.x;
    atomicAdd(&accum[(long)dstI[e] * HDIM + h], m_out[e * HDIM + h]);
}

// ---------------------------------------------------------------------------
// Main per-level GRU kernel. Block = 16 edge-rows, 256 threads = 8 waves.
// Each wave computes 2 N-tiles (32 cols); 8 waves cover all 256 cols.
// ---------------------------------------------------------------------------
__global__ __launch_bounds__(256)
void level_gemm_kernel(const int* __restrict__ eids,
                       const int* __restrict__ srcI, const int* __restrict__ dstI,
                       const _Float16* __restrict__ x16,
                       const float* __restrict__ s_buf, const float* __restrict__ rm_buf,
                       const _Float16* __restrict__ wz, const _Float16* __restrict__ wh,
                       const _Float16* __restrict__ wr, const _Float16* __restrict__ ur,
                       const float* __restrict__ bz, const float* __restrict__ bh,
                       const float* __restrict__ bu,
                       float* __restrict__ m_out, _Float16* __restrict__ r16) {
    __shared__ __attribute__((aligned(16))) _Float16 SX[16 * LDSH];
    __shared__ __attribute__((aligned(16))) _Float16 DX[16 * LDSH];
    __shared__ __attribute__((aligned(16))) _Float16 SS[16 * LDSH];
    __shared__ __attribute__((aligned(16))) _Float16 RM[16 * LDSH];
    __shared__ __attribute__((aligned(16))) _Float16 ML[16 * LDSH];
    __shared__ int EID[16];

    const int tid = threadIdx.x;
    const int rowbase = blockIdx.x * 16;
    if (tid < 16) EID[tid] = eids[rowbase + tid];
    __syncthreads();

    // Stage A operands (f16) into LDS.
    for (int idx = tid; idx < 16 * HDIM; idx += 256) {
        int i = idx >> 8;
        int h = idx & 255;
        int e = EID[i];
        int sn = srcI[e], dn = dstI[e];
        SX[i * LDSH + h] = x16[(long)sn * HDIM + h];
        DX[i * LDSH + h] = x16[(long)dn * HDIM + h];
        SS[i * LDSH + h] = (_Float16)s_buf[(long)(rowbase + i) * HDIM + h];
        RM[i * LDSH + h] = (_Float16)rm_buf[(long)(rowbase + i) * HDIM + h];
    }
    __syncthreads();

    const int w = tid >> 5;
    const int l = tid & 31;
    const int mrow = l & 15;
    const _Float16* Asx = SX + mrow * LDSH;
    const _Float16* Ass = SS + mrow * LDSH;
    const _Float16* Arm = RM + mrow * LDSH;
    const _Float16* Adx = DX + mrow * LDSH;

    v8f accz[2], acch[2], accr[2];

    // Phase 1: z-pre = [sx,s]@Wz ; h-pre = [sx,rm]@Wh ; r-partial = dx@Wr
    for (int p = 0; p < 2; ++p) {
        const int nt = 2 * w + p;
        v8f az = {}; v8f ah = {}; v8f ar = {};
#pragma unroll
        for (int kt = 0; kt < 16; ++kt) {
            v16h a1 = (kt < 8) ? loadA(Asx, kt * 32, l) : loadA(Ass, (kt - 8) * 32, l);
            v16h a2 = (kt < 8) ? a1 : loadA(Arm, (kt - 8) * 32, l);
            az = wmma_f16(a1, loadB(wz, kt, nt, l), az);
            ah = wmma_f16(a2, loadB(wh, kt, nt, l), ah);
            if (kt < 8) {
                ar = wmma_f16(loadA(Adx, kt * 32, l), loadB(wr, kt, nt, l), ar);
            }
        }
        accz[p] = az; acch[p] = ah; accr[p] = ar;
    }

    // Phase 2: gates + m_lvl; write m to d_out and f16 copy to LDS for Ur GEMM.
    for (int p = 0; p < 2; ++p) {
        const int nt = 2 * w + p;
        const int ncol = nt * 16 + (l & 15);
        const float bzv = bz[ncol];
        const float bhv = bh[ncol];
#pragma unroll
        for (int r = 0; r < 8; ++r) {
            int m = r + ((l < 16) ? 0 : 8);   // C/D layout: VGPR r -> M=r (+8 for hi half-wave)
            float z  = sigmoidf_(accz[p][r] + bzv);
            float mc = tanhf(acch[p][r] + bhv);
            float sv = s_buf[(long)(rowbase + m) * HDIM + ncol];
            float mv = (1.0f - z) * sv + z * mc;
            m_out[(long)EID[m] * HDIM + ncol] = mv;
            ML[m * LDSH + ncol] = (_Float16)mv;
        }
    }
    __syncthreads();   // all 256 columns of m_lvl must be staged before Ur K-loop

    // Phase 3: r = sigmoid(dx@Wr + m_lvl@Ur + b)
    const _Float16* Aml = ML + mrow * LDSH;
    for (int p = 0; p < 2; ++p) {
        const int nt = 2 * w + p;
        v8f ar = accr[p];
#pragma unroll
        for (int kt = 0; kt < 8; ++kt) {
            ar = wmma_f16(loadA(Aml, kt * 32, l), loadB(ur, kt, nt, l), ar);
        }
        const int ncol = nt * 16 + (l & 15);
        const float buv = bu[ncol];
#pragma unroll
        for (int r = 0; r < 8; ++r) {
            int m = r + ((l < 16) ? 0 : 8);
            float rv = sigmoidf_(ar[r] + buv);
            r16[(long)EID[m] * HDIM + ncol] = (_Float16)rv;
        }
    }
}

// ---------------------------------------------------------------------------
// Final readout: root_vecs = relu([x, accum] @ Wg + bg) for the 256 roots only.
// ---------------------------------------------------------------------------
__global__ __launch_bounds__(256)
void final_gemm_kernel(const int* __restrict__ root_ids,
                       const _Float16* __restrict__ x16, const float* __restrict__ accum,
                       const _Float16* __restrict__ wg, const float* __restrict__ bg,
                       float* __restrict__ out) {
    __shared__ __attribute__((aligned(16))) _Float16 AX[16 * LDSH];
    __shared__ __attribute__((aligned(16))) _Float16 AA[16 * LDSH];
    __shared__ int NID[16];

    const int tid = threadIdx.x;
    const int rowbase = blockIdx.x * 16;
    if (tid < 16) NID[tid] = root_ids[rowbase + tid];
    __syncthreads();

    for (int idx = tid; idx < 16 * HDIM; idx += 256) {
        int i = idx >> 8;
        int h = idx & 255;
        int n = NID[i];
        AX[i * LDSH + h] = x16[(long)n * HDIM + h];
        AA[i * LDSH + h] = (_Float16)accum[(long)n * HDIM + h];
    }
    __syncthreads();

    const int w = tid >> 5;
    const int l = tid & 31;
    const int mrow = l & 15;
    for (int p = 0; p < 2; ++p) {
        const int nt = 2 * w + p;
        v8f acc = {};
#pragma unroll
        for (int kt = 0; kt < 16; ++kt) {
            v16h a = (kt < 8) ? loadA(AX + mrow * LDSH, kt * 32, l)
                              : loadA(AA + mrow * LDSH, (kt - 8) * 32, l);
            acc = wmma_f16(a, loadB(wg, kt, nt, l), acc);
        }
        const int ncol = nt * 16 + (l & 15);
        const float bv = bg[ncol];
#pragma unroll
        for (int r = 0; r < 8; ++r) {
            int m = r + ((l < 16) ? 0 : 8);
            float v = acc[r] + bv;
            out[(long)(rowbase + m) * HDIM + ncol] = (v > 0.0f) ? v : 0.0f;
        }
    }
}

// ---------------------------------------------------------------------------
// Host launcher
// ---------------------------------------------------------------------------
extern "C" void kernel_launch(void* const* d_in, const int* in_sizes, int n_in,
                              void* d_out, int out_size, void* d_ws, size_t ws_size,
                              hipStream_t stream) {
    const float* emb  = (const float*)d_in[0];
    const float* Wz_w = (const float*)d_in[1];
    const float* Wz_b = (const float*)d_in[2];
    const float* Wr_w = (const float*)d_in[3];
    const float* Ur_w = (const float*)d_in[4];
    const float* Ur_b = (const float*)d_in[5];
    const float* Wh_w = (const float*)d_in[6];
    const float* Wh_b = (const float*)d_in[7];
    const float* Wg_w = (const float*)d_in[8];
    const float* Wg_b = (const float*)d_in[9];
    const int* wid      = (const int*)d_in[10];
    const int* srcI     = (const int*)d_in[11];
    const int* dstI     = (const int*)d_in[12];
    const int* root_ids = (const int*)d_in[13];

    // Workspace layout (bytes)
    char* ws = (char*)d_ws;
    _Float16* wz16 = (_Float16*)(ws + 0);          // 512*256*2 = 262144
    _Float16* wh16 = (_Float16*)(ws + 262144);     // 262144
    _Float16* wg16 = (_Float16*)(ws + 524288);     // 262144
    _Float16* wr16 = (_Float16*)(ws + 786432);     // 131072
    _Float16* ur16 = (_Float16*)(ws + 917504);     // 131072
    _Float16* x16  = (_Float16*)(ws + 1048576);                 // N*H*2  = 33423360
    _Float16* r16  = (_Float16*)(ws + 34471936);                // E*H*2  = 66584576
    float*    sbuf = (float*)   (ws + 101056512);               // 32768*H*4 = 33554432
    float*    rmb  = (float*)   (ws + 134610944);               // 32768*H*4 = 33554432
    float*    accum = sbuf;  // reused after the level loop (needs N*H*4 = 66.8MB)

    float* m_out    = (float*)d_out;                        // [E, H]
    float* root_out = (float*)d_out + (long)NEDGES * HDIM;  // [T, H]

    // 1) Pre-swizzle weights to f16 WMMA-fragment layout.
    cvt_frag_kernel<<<512, 256, 0, stream>>>(Wz_w, wz16, 512);
    cvt_frag_kernel<<<512, 256, 0, stream>>>(Wh_w, wh16, 512);
    cvt_frag_kernel<<<512, 256, 0, stream>>>(Wg_w, wg16, 512);
    cvt_frag_kernel<<<256, 256, 0, stream>>>(Wr_w, wr16, 256);
    cvt_frag_kernel<<<256, 256, 0, stream>>>(Ur_w, ur16, 256);

    // 2) Node embeddings -> f16.
    gather_x_kernel<<<NNODES, 256, 0, stream>>>(emb, wid, x16);

    // 3) 14 sequential BFS levels.
    for (int lvl = 0; lvl < NLEVELS; ++lvl) {
        const int* eids = (const int*)d_in[14 + lvl];
        const int* lgs  = (const int*)d_in[28 + lvl];
        const int* lgd  = (const int*)d_in[42 + lvl];
        const int rows = in_sizes[14 + lvl];   // k * T (multiple of 16)
        const int len  = in_sizes[28 + lvl];   // predecessor-list length

        zerof_kernel<<<rows, 256, 0, stream>>>(sbuf, (long)rows * HDIM);
        zerof_kernel<<<rows, 256, 0, stream>>>(rmb,  (long)rows * HDIM);
        if (len > 0) {
            scatter_sm_kernel<<<len, 256, 0, stream>>>(lgs, lgd, m_out, r16, sbuf, rmb);
        }
        level_gemm_kernel<<<rows / 16, 256, 0, stream>>>(
            eids, srcI, dstI, x16, sbuf, rmb,
            wz16, wh16, wr16, ur16, Wz_b, Wh_b, Ur_b, m_out, r16);
    }

    // 4) Node accumulation + root readout.
    zerof_kernel<<<NNODES, 256, 0, stream>>>(accum, (long)NNODES * HDIM);
    scatter_accum_kernel<<<NEDGES, 256, 0, stream>>>(dstI, m_out, accum);
    final_gemm_kernel<<<NTREES / 16, 256, 0, stream>>>(root_ids, x16, accum, wg16, Wg_b, root_out);
}
